// Attention_7464653160631
// MI455X (gfx1250) — matrix operand order repaired and checked by reading, not compiled, not run on hardware
//
#include <hip/hip_runtime.h>
#include <hip/hip_bf16.h>
#include <stdint.h>

// ---------------- problem constants ----------------
#define B_      64
#define T_      2048
#define D_LSTM  1024
#define D_ENC   512
#define D_ATT   128
#define N_F     32
#define KW      31
#define PADW    15
#define TC      256              // t-values per chunk (per block)
#define NCHUNK  (T_ / TC)        // 8 chunks per batch
#define ROWE    528              // LDS tile row stride in bf16 elements (avoid bank conflicts)

typedef __attribute__((ext_vector_type(16))) __bf16 v16bf;
typedef __attribute__((ext_vector_type(8)))  float  v8f;
typedef __attribute__((ext_vector_type(4)))  float  v4f;   // native vector for NT loads

struct Raw32 { uint4 a, b; };    // 32 bytes, trivially copyable

__device__ __forceinline__ uint16_t f2bf(float x) {
    uint32_t u = __builtin_bit_cast(uint32_t, x);
    uint32_t r = u + 0x7FFFu + ((u >> 16) & 1u);   // round-to-nearest-even
    return (uint16_t)(r >> 16);
}
__device__ __forceinline__ float bf2f_lo(uint32_t p) {
    return __builtin_bit_cast(float, p << 16);
}
__device__ __forceinline__ float bf2f_hi(uint32_t p) {
    return __builtin_bit_cast(float, p & 0xFFFF0000u);
}
// Build a 16-element bf16 WMMA fragment from two 16-byte chunks.
__device__ __forceinline__ v16bf load_frag(const uint16_t* p0, const uint16_t* p1) {
    Raw32 r;
    r.a = *(const uint4*)p0;
    r.b = *(const uint4*)p1;
    return __builtin_bit_cast(v16bf, r);
}

// ---------------- kernel 1: pack Wm/Wloc to bf16 ----------------
__global__ void pack_weights(const float* __restrict__ Wm, const float* __restrict__ Wloc,
                             uint16_t* __restrict__ wmB, uint16_t* __restrict__ wlB) {
    int i = blockIdx.x * 256 + threadIdx.x;
    if (i < D_ATT * D_ENC) {
        wmB[i] = f2bf(Wm[i]);                 // [a][d], d (=K) contiguous: B-frag friendly
    } else {
        int j = i - D_ATT * D_ENC;
        if (j < D_ATT * N_F) wlB[j] = f2bf(Wloc[j]);   // [a][f], f (=K) contiguous
    }
}

// ---------------- kernel 2: qb[b,a] = query.Wq + bq + bm (folded bias) ----------------
__global__ __launch_bounds__(128, 1)
void qproj(const float* __restrict__ query, const float* __restrict__ Wq,
           const float* __restrict__ bq, const float* __restrict__ bm,
           float* __restrict__ qb) {
    int b = blockIdx.x, a = threadIdx.x;              // 128 threads: one output each
    const float* q = query + (size_t)b * D_LSTM;
    const float* w = Wq    + (size_t)a * D_LSTM;
    float acc = bq[a] + bm[a];
    for (int d = 0; d < D_LSTM; d += 4) {
        float4 qq = *(const float4*)(q + d);
        float4 ww = *(const float4*)(w + d);
        acc += qq.x * ww.x + qq.y * ww.y + qq.z * ww.z + qq.w * ww.w;
    }
    qb[b * D_ATT + a] = acc;
}

// ---------------- kernel 3: location conv1d -> bf16 [B,T,32] ----------------
__global__ __launch_bounds__(256, 1)
void loc_conv(const float* __restrict__ attw, const float* __restrict__ Wconv,
              uint16_t* __restrict__ convA) {
    __shared__ float aw[2][TC + KW - 1];      // haloed window, zero padded
    __shared__ float wc[N_F * 2 * KW];
    int b = blockIdx.y, t0 = blockIdx.x * TC, tid = threadIdx.x;

    for (int i = tid; i < 2 * (TC + KW - 1); i += 256) {
        int c = i / (TC + KW - 1), j = i % (TC + KW - 1);
        int t = t0 + j - PADW;
        aw[c][j] = (t >= 0 && t < T_) ? attw[((size_t)b * 2 + c) * T_ + t] : 0.f;
    }
    for (int i = tid; i < N_F * 2 * KW; i += 256) wc[i] = Wconv[i];
    __syncthreads();

    uint32_t out[N_F / 2];
    for (int f = 0; f < N_F; ++f) {
        float acc = 0.f;
        for (int c = 0; c < 2; ++c) {
#pragma unroll
            for (int j = 0; j < KW; ++j)
                acc += aw[c][tid + j] * wc[(f * 2 + c) * KW + j];
        }
        uint16_t h = f2bf(acc);
        if (f & 1) out[f >> 1] |= ((uint32_t)h) << 16; else out[f >> 1] = h;
    }
    uint32_t* dst = (uint32_t*)(convA + ((size_t)b * T_ + t0 + tid) * N_F);
#pragma unroll
    for (int k = 0; k < N_F / 2; ++k) dst[k] = out[k];
}

// ---------------- kernel 4: fused k-proj + loc + tanh-energy + online softmax + ctx ----------------
__global__ __launch_bounds__(256, 1)
void attn_main(const float* __restrict__ mem, const float* __restrict__ Wv,
               const uint16_t* __restrict__ wm, const uint16_t* __restrict__ wloc,
               const float* __restrict__ qb, const uint16_t* __restrict__ convA,
               float* __restrict__ eOut, float* __restrict__ ml, float* __restrict__ ctxp) {
    const int b     = blockIdx.y;
    const int chunk = blockIdx.x;
    const int tid   = threadIdx.x;
    const int wave  = tid >> 5;            // 0..7, each owns an N-tile of 16
    const int lane  = tid & 31;
    const int hi    = lane >> 4;           // lane half selects K sub-block (WMMA layout)
    const int l15   = lane & 15;

    __shared__ uint16_t tileA[16 * ROWE];  // 16 x 512 bf16 memory tile (padded rows)
    __shared__ float    e_tile[16];
    __shared__ float    ctxs[D_ENC];
    __shared__ float    mrun_s, lrun_s;

    for (int d = tid; d < D_ENC; d += 256) ctxs[d] = 0.f;
    if (tid == 0) { mrun_s = -3.0e38f; lrun_s = 0.f; }

    // Preload register-resident B fragments: Wm (16 K-steps) + Wloc (1 step), bf16.
    // B layout (32x16, 16-bit): lane n=l15 holds column n; lanes 0-15 K 0-7 & 16-23,
    // lanes 16-31 K 8-15 & 24-31 -> two 16B loads per lane.
    const int n0 = wave * 16;
    v16bf bfr[16];
#pragma unroll
    for (int s = 0; s < 16; ++s) {
        const uint16_t* p = wm + (size_t)(n0 + l15) * D_ENC + s * 32 + hi * 8;
        bfr[s] = load_frag(p, p + 16);
    }
    v16bf blf;
    {
        const uint16_t* p = wloc + (size_t)(n0 + l15) * N_F + hi * 8;
        blf = load_frag(p, p + 16);
    }
    const float vv = Wv[n0 + l15];                       // Wv column for this lane's N
    const float qv = qb[b * D_ATT + n0 + l15];           // C init: q + bq + bm (N-dependent only)

    const int t0chunk = chunk * TC;
    for (int tt = 0; tt < TC / 16; ++tt) {
        const int t0 = t0chunk + tt * 16;
        __syncthreads();   // previous iteration fully consumed tileA / mrun

        // ---- cooperative load: 16 x 512 fp32 -> bf16 in LDS (non-temporal stream) ----
        for (int g = tid; g < 16 * 128; g += 256) {      // 2048 float4 groups
            int r = g >> 7, c4 = g & 127;
            const v4f* gp =
                (const v4f*)(mem + ((size_t)b * T_ + (t0 + r)) * D_ENC + c4 * 4);
            const v4f f4 = __builtin_nontemporal_load(gp);
            uint2 pk;
            pk.x = (uint32_t)f2bf(f4.x) | ((uint32_t)f2bf(f4.y) << 16);
            pk.y = (uint32_t)f2bf(f4.z) | ((uint32_t)f2bf(f4.w) << 16);
            *(uint2*)(&tileA[r * ROWE + c4 * 4]) = pk;
        }
        if (tid < 16) e_tile[tid] = 0.f;
        __syncthreads();

        // prefetch next tile into cache (global_prefetch_b8)
        if (tt + 1 < TC / 16) {
            const float* np = mem + ((size_t)b * T_ + (t0 + 16)) * D_ENC;
            __builtin_prefetch(np + tid * 32, 0, 3);
        }

        // ---- WMMA: acc = qb + memTile(bf16) @ Wm(bf16) + convA(bf16) @ Wloc(bf16) ----
        v8f acc;
#pragma unroll
        for (int i = 0; i < 8; ++i) acc[i] = qv;

#pragma unroll
        for (int s = 0; s < 16; ++s) {
            // A layout (16x32, 16-bit): lane row r=l15; lanes 0-15 hold K 0-7 & 16-23,
            // lanes 16-31 hold K 8-15 & 24-31 -> two 16B LDS loads.
            const uint16_t* lp = &tileA[l15 * ROWE + s * 32 + hi * 8];
            v16bf afr = load_frag(lp, lp + 16);
            acc = __builtin_amdgcn_wmma_f32_16x16x32_bf16(false, afr, false, bfr[s],
                                                          (short)0, acc, false, false);
        }
        {   // location term, A pulled straight from global (row-major [t][32] bf16)
            const uint16_t* cp = convA + ((size_t)b * T_ + t0 + l15) * N_F + hi * 8;
            v16bf afr = load_frag(cp, cp + 16);
            acc = __builtin_amdgcn_wmma_f32_16x16x32_bf16(false, afr, false, blf,
                                                          (short)0, acc, false, false);
        }

        // ---- energies: e[m] = sum_n Wv[n] * tanh(acc[m,n]) ----
        // C layout: VGPR i -> row m = i + 8*hi; lane l15 -> column n.
#pragma unroll
        for (int i = 0; i < 8; ++i) {
            float tv = tanhf(acc[i]) * vv;
            tv += __shfl_xor(tv, 1, 16);
            tv += __shfl_xor(tv, 2, 16);
            tv += __shfl_xor(tv, 4, 16);
            tv += __shfl_xor(tv, 8, 16);
            if (l15 == 0) atomicAdd(&e_tile[i + hi * 8], tv);
        }
        __syncthreads();   // e_tile complete across all 8 waves

        // ---- online softmax + context accumulation (single pass over memory) ----
        float mold = mrun_s, lold = lrun_s;
        float ee[16];
        float tmax = mold;
#pragma unroll
        for (int m = 0; m < 16; ++m) { ee[m] = e_tile[m]; tmax = fmaxf(tmax, ee[m]); }
        float scale = __expf(mold - tmax);
        float w16[16], lsum = 0.f;
#pragma unroll
        for (int m = 0; m < 16; ++m) { w16[m] = __expf(ee[m] - tmax); lsum += w16[m]; }

        {   // two consecutive d per thread -> one packed ds_load_b32 per m
            const int d0 = tid * 2;
            float a0 = ctxs[d0] * scale;
            float a1 = ctxs[d0 + 1] * scale;
#pragma unroll
            for (int m = 0; m < 16; ++m) {
                uint32_t p = *(const uint32_t*)(&tileA[m * ROWE + d0]);
                a0 += w16[m] * bf2f_lo(p);
                a1 += w16[m] * bf2f_hi(p);
            }
            ctxs[d0]     = a0;
            ctxs[d0 + 1] = a1;
        }
        if (tid < 16) eOut[(size_t)b * T_ + t0 + tid] = e_tile[tid];   // bv cancels in softmax
        __syncthreads();   // all threads done with mrun/lrun & tileA reads
        if (tid == 0) { mrun_s = tmax; lrun_s = lold * scale + lsum; }
    }
    __syncthreads();

    // ---- chunk partials: (m, l, ctx numerator) ----
    const int idx = b * NCHUNK + chunk;
    if (tid == 0) { ml[idx * 2] = mrun_s; ml[idx * 2 + 1] = lrun_s; }
    for (int d = tid; d < D_ENC; d += 256) ctxp[(size_t)idx * D_ENC + d] = ctxs[d];
}

// ---------------- kernel 5: combine chunk partials, emit outputs ----------------
__global__ __launch_bounds__(256, 1)
void attn_finalize(const float* __restrict__ e, const float* __restrict__ ml,
                   const float* __restrict__ ctxp,
                   float* __restrict__ outCtx, float* __restrict__ outW) {
    int b = blockIdx.x, tid = threadIdx.x;
    float mi[NCHUNK], li[NCHUNK], M = -3.0e38f;
#pragma unroll
    for (int i = 0; i < NCHUNK; ++i) {
        mi[i] = ml[(b * NCHUNK + i) * 2];
        li[i] = ml[(b * NCHUNK + i) * 2 + 1];
        M = fmaxf(M, mi[i]);
    }
    float L = 0.f, sc[NCHUNK];
#pragma unroll
    for (int i = 0; i < NCHUNK; ++i) { sc[i] = __expf(mi[i] - M); L += li[i] * sc[i]; }
    const float invL = 1.f / L;

    for (int d = tid; d < D_ENC; d += 256) {
        float a = 0.f;
#pragma unroll
        for (int i = 0; i < NCHUNK; ++i)
            a += ctxp[((size_t)(b * NCHUNK + i)) * D_ENC + d] * sc[i];
        outCtx[(size_t)b * D_ENC + d] = a * invL;
    }
    for (int t = tid; t < T_; t += 256)
        outW[(size_t)b * T_ + t] = __expf(e[(size_t)b * T_ + t] - M) * invL;
}

// ---------------- host launcher ----------------
extern "C" void kernel_launch(void* const* d_in, const int* in_sizes, int n_in,
                              void* d_out, int out_size, void* d_ws, size_t ws_size,
                              hipStream_t stream) {
    (void)in_sizes; (void)n_in; (void)out_size; (void)ws_size;
    const float* query  = (const float*)d_in[0];
    const float* memory = (const float*)d_in[1];
    const float* attw   = (const float*)d_in[2];
    const float* Wq     = (const float*)d_in[3];
    const float* bq     = (const float*)d_in[4];
    const float* Wm     = (const float*)d_in[5];
    const float* bm     = (const float*)d_in[6];
    const float* Wconv  = (const float*)d_in[7];
    const float* Wloc   = (const float*)d_in[8];
    const float* Wv     = (const float*)d_in[9];
    // d_in[10] = bv: additive scalar before softmax -> cancels; unused.

    char* ws = (char*)d_ws;
    size_t o = 0;
    uint16_t* wmB   = (uint16_t*)(ws + o); o += (size_t)D_ATT * D_ENC * 2;      // 128 KB
    uint16_t* wlB   = (uint16_t*)(ws + o); o += (size_t)D_ATT * N_F * 2;        // 8 KB
    float*    qb    = (float*)   (ws + o); o += (size_t)B_ * D_ATT * 4;         // 32 KB
    uint16_t* convA = (uint16_t*)(ws + o); o += (size_t)B_ * T_ * N_F * 2;      // 8 MB
    float*    eBuf  = (float*)   (ws + o); o += (size_t)B_ * T_ * 4;            // 512 KB
    float*    ml    = (float*)   (ws + o); o += (size_t)B_ * NCHUNK * 2 * 4;    // 4 KB
    float*    ctxp  = (float*)   (ws + o); o += (size_t)B_ * NCHUNK * D_ENC * 4;// 1 MB

    float* outCtx = (float*)d_out;                 // [B, 512]
    float* outW   = outCtx + (size_t)B_ * D_ENC;   // [B, 2048]

    int packN = (D_ATT * D_ENC + D_ATT * N_F + 255) / 256;
    pack_weights<<<packN, 256, 0, stream>>>(Wm, Wloc, wmB, wlB);
    qproj<<<B_, 128, 0, stream>>>(query, Wq, bq, bm, qb);
    loc_conv<<<dim3(T_ / TC, B_), 256, 0, stream>>>(attw, Wconv, convA);
    attn_main<<<dim3(NCHUNK, B_), 256, 0, stream>>>(memory, Wv, wmB, wlB, qb, convA,
                                                    eBuf, ml, ctxp);
    attn_finalize<<<B_, 256, 0, stream>>>(eBuf, ml, ctxp, outCtx, outW);
}
